// _Decoder_34978213658797
// MI455X (gfx1250) — compile-verified, matrix operand order checked by reading, not run on hardware
//
#include <hip/hip_runtime.h>
#include <math.h>

// ---------------------------------------------------------------------------
// CDNA5 (gfx1250) per-layer WMMA GEMM for the 23-head MLP decoder.
// wave32, v_wmma_f32_16x16x32_bf16, fp32 accumulation, double-buffered LDS,
// async global->LDS staging for the activation tile (ASYNCcnt path).
// ---------------------------------------------------------------------------

typedef __attribute__((ext_vector_type(16))) __bf16 v16bf;
typedef __attribute__((ext_vector_type(8)))  float  v8f;

#define BM 128
#define BN 128
#define BK 32
#define LDSTR 40   // 32 + 8 pad elems -> 80B rows (16B aligned, bank-spread)

union FragBF { unsigned int u[8]; v16bf v; };
union Chunk  { unsigned short s[8]; uint4 q; };

static __device__ __forceinline__ unsigned int bf16_bits(float f) {
  return (unsigned int)__builtin_bit_cast(unsigned short, (__bf16)f);
}

// Generic pointers to LDS carry the LDS byte offset in their low 32 bits
// (flat->LDS mapping truncates to addr[31:0]).
static __device__ __forceinline__ unsigned int lds_addr(const void* p) {
  return (unsigned int)(unsigned long long)p;
}

// ---- tile staging --------------------------------------------------------

// A tile, all 32 K-columns in range: 2 async 16B global->LDS copies per
// thread (no VGPR round-trip, tracked by ASYNCcnt).
static __device__ __forceinline__
void stage_a_full(const __bf16* A, int lda, int rowTile, int k0,
                  int tid, __bf16 (&As)[BM][LDSTR]) {
#pragma unroll
  for (int p = 0; p < 2; ++p) {
    int r  = (tid >> 2) + p * 64;
    int kq = (tid & 3) * 8;
    unsigned long long ga =
        (unsigned long long)(A + (size_t)(rowTile + r) * lda + k0 + kq);
    unsigned int la = lds_addr(&As[r][kq]);
    asm volatile("global_load_async_to_lds_b128 %0, %1, off"
                 :: "v"(la), "v"(ga) : "memory");
  }
}

// A tile, ragged K edge (runs at most once per block): manual guarded copy.
static __device__ __forceinline__
void stage_a_edge(const __bf16* __restrict__ A, int lda, int rowTile, int k0,
                  int K, int tid, __bf16 (&As)[BM][LDSTR]) {
#pragma unroll
  for (int p = 0; p < 2; ++p) {
    int r  = (tid >> 2) + p * 64;
    int kq = (tid & 3) * 8;
    const __bf16* src = A + (size_t)(rowTile + r) * lda + k0 + kq;
    Chunk ch;
#pragma unroll
    for (int e = 0; e < 8; ++e) {
      int kk = k0 + kq + e;
      ch.s[e] = (kk < K) ? __builtin_bit_cast(unsigned short, src[e])
                         : (unsigned short)0;
    }
    *(uint4*)&As[r][kq] = ch.q;
  }
}

// W tile transposed [n][k], no guards: 16 coalesced b32 loads,
// fp32->bf16 pairs packed into 8 ds_store_b32 per thread.
// Also prefetches the tile two K-steps ahead into L2.
static __device__ __forceinline__
void stage_w_full(const float* __restrict__ W, int K, int N, int colTile,
                  int k0, int tid, __bf16 (&Ws)[BN][LDSTR]) {
  if (k0 + 2 * BK < K)   // warm L2 ahead of the double buffer
    __builtin_prefetch(W + (size_t)(k0 + 2 * BK + (tid >> 7)) * N +
                           colTile + (tid & 127), 0, 1);
#pragma unroll
  for (int i = 0; i < 8; ++i) {
    int e  = i * 256 + tid;   // 0..2047 over [16 kpair][128 n]
    int kp = e >> 7;
    int nn = e & 127;
    const float* p = W + (size_t)(k0 + 2 * kp) * N + colTile + nn;
    unsigned int lo = bf16_bits(p[0]);
    unsigned int hi = bf16_bits(p[N]);
    *(unsigned int*)&Ws[nn][2 * kp] = lo | (hi << 16);
  }
}

// W tile transposed with K/N guards (K-edge iteration or last N block).
static __device__ __forceinline__
void stage_w_guard(const float* __restrict__ W, int K, int N, int colTile,
                   int k0, int tid, __bf16 (&Ws)[BN][LDSTR]) {
#pragma unroll
  for (int i = 0; i < 8; ++i) {
    int e  = i * 256 + tid;
    int kp = e >> 7;
    int nn = e & 127;
    int gk = k0 + 2 * kp;
    int gn = colTile + nn;
    unsigned int lo = 0, hi = 0;
    if (gn < N) {
      if (gk < K)     lo = bf16_bits(W[(size_t)gk * N + gn]);
      if (gk + 1 < K) hi = bf16_bits(W[(size_t)(gk + 1) * N + gn]);
    }
    *(unsigned int*)&Ws[nn][2 * kp] = lo | (hi << 16);
  }
}

// ---- wmma compute over one staged tile -----------------------------------

static __device__ __forceinline__
void wmma_tile(const __bf16 (&As)[BM][LDSTR], const __bf16 (&Ws)[BN][LDSTR],
               int wm, int wn, int lane, v8f (&acc)[4][2]) {
  const int kh  = (lane >> 4) * 8;
  const int l15 = lane & 15;
  FragBF bf[2];
#pragma unroll
  for (int j = 0; j < 2; ++j) {
    int c = wn + j * 16 + l15;
    *(uint4*)&bf[j].u[0] = *(const uint4*)&Ws[c][kh];
    *(uint4*)&bf[j].u[4] = *(const uint4*)&Ws[c][16 + kh];
  }
#pragma unroll
  for (int i = 0; i < 4; ++i) {
    FragBF af;
    int r = wm + i * 16 + l15;
    *(uint4*)&af.u[0] = *(const uint4*)&As[r][kh];
    *(uint4*)&af.u[4] = *(const uint4*)&As[r][16 + kh];
#pragma unroll
    for (int j = 0; j < 2; ++j)
      acc[i][j] = __builtin_amdgcn_wmma_f32_16x16x32_bf16(
          false, af.v, false, bf[j].v, (short)0, acc[i][j], false, false);
  }
}

// ---- kernels -------------------------------------------------------------

// concat = [x | one_hot(cond)] as bf16, row stride 264 (528B, 16B aligned)
__global__ void __launch_bounds__(264)
prep_concat_kernel(const float* __restrict__ x, const int* __restrict__ cond,
                   __bf16* __restrict__ out) {
  int r = blockIdx.x;
  int c = threadIdx.x;
  if (c >= 264) return;
  float v;
  if (c < 256) v = x[r * 256 + c];
  else         v = (cond[r] == (c - 256)) ? 1.0f : 0.0f;
  out[r * 264 + c] = (__bf16)v;
}

// C[1024 x N] = act(A[1024 x K] @ W[K x N] + bias)
// FINAL=false: ReLU -> bf16 outH (row stride ldpad). FINAL=true: sigmoid -> fp32 outF.
template <bool FINAL>
__global__ void __launch_bounds__(256)
mlp_gemm_kernel(const __bf16* __restrict__ A, int lda, int K,
                const float* __restrict__ W, const float* __restrict__ bias,
                int N, int ldpad,
                __bf16* __restrict__ outH,
                float* __restrict__ outF, int ldout, int colOff) {
  __shared__ __bf16 As[2][BM][LDSTR];
  __shared__ __bf16 Ws[2][BN][LDSTR];

  const int tid  = threadIdx.x;
  const int lane = tid & 31;
  const int wave = tid >> 5;
  const int wm = (wave >> 2) * 64;     // wave row base within block tile
  const int wn = (wave & 3) * 32;      // wave col base within block tile
  const int rowTile = blockIdx.x * BM;
  const int colTile = blockIdx.y * BN;
  const bool fullN = (colTile + BN) <= N;   // block-uniform

  const int nFull = K >> 5;
  const int nIter = nFull + ((K & 31) ? 1 : 0);

  v8f acc[4][2] = {};

  // prologue: stage tile 0 into buffer 0
  if (0 < nFull) {
    stage_a_full(A, lda, rowTile, 0, tid, As[0]);
    if (fullN) stage_w_full(W, K, N, colTile, 0, tid, Ws[0]);
    else       stage_w_guard(W, K, N, colTile, 0, tid, Ws[0]);
  } else {
    stage_a_edge(A, lda, rowTile, 0, K, tid, As[0]);
    stage_w_guard(W, K, N, colTile, 0, tid, Ws[0]);
  }

  for (int it = 0; it < nIter; ++it) {
    // async deposits into the buffer being published must be complete
    asm volatile("s_wait_asynccnt 0" ::: "memory");
    __syncthreads();                       // publish buffer it&1
    const int buf = it & 1;
    if (it + 1 < nIter) {                  // stage next tile into other buffer
      const int nxt = buf ^ 1;
      const int k0 = (it + 1) * BK;
      if (it + 1 < nFull) {
        stage_a_full(A, lda, rowTile, k0, tid, As[nxt]);
        if (fullN) stage_w_full(W, K, N, colTile, k0, tid, Ws[nxt]);
        else       stage_w_guard(W, K, N, colTile, k0, tid, Ws[nxt]);
      } else {
        stage_a_edge(A, lda, rowTile, k0, K, tid, As[nxt]);
        stage_w_guard(W, K, N, colTile, k0, tid, Ws[nxt]);
      }
    }
    wmma_tile(As[buf], Ws[buf], wm, wn, lane, acc);
  }

  // ---- epilogue: C/D layout -> lane (M half, N), VGPR r -> row ----
  const int laneN = lane & 15;
  const int mh    = (lane >> 4) * 8;
#pragma unroll
  for (int i = 0; i < 4; ++i) {
#pragma unroll
    for (int j = 0; j < 2; ++j) {
      int col = colTile + wn + j * 16 + laneN;
#pragma unroll
      for (int r8 = 0; r8 < 8; ++r8) {
        int row = rowTile + wm + i * 16 + mh + r8;
        float v = acc[i][j][r8];
        if constexpr (FINAL) {
          if (col < N) {
            float t = v + bias[col];
            outF[(size_t)row * ldout + colOff + col] = 1.0f / (1.0f + __expf(-t));
          }
        } else {
          if (col < ldpad) {   // also zero the <8-wide alignment pad
            float t = (col < N) ? fmaxf(v + bias[col], 0.0f) : 0.0f;
            outH[(size_t)row * ldpad + col] = (__bf16)t;
          }
        }
      }
    }
  }
}

// ---------------------------------------------------------------------------
// Host side
// ---------------------------------------------------------------------------

static int isqrt_floor(long v) {
  long r = (long)sqrt((double)v);
  while ((r + 1) * (r + 1) <= v) ++r;
  while (r * r > v) --r;
  return (int)r;
}

extern "C" void kernel_launch(void* const* d_in, const int* in_sizes, int n_in,
                              void* d_out, int out_size, void* d_ws, size_t ws_size,
                              hipStream_t stream) {
  static const int SIZES[23] = {8000, 7700, 7400, 7100, 6800, 6500, 6200, 5900,
                                5600, 5300, 5000, 4700, 4400, 4100, 3800, 3500,
                                3200, 2900, 2600, 2300, 2000, 1700, 1400};
  const int BATCH = 1024;
  const int DIN   = 264;       // 256 + 8 one-hot
  const int OUT_TOTAL = 108100;

  const float* x    = (const float*)d_in[0];
  const int*   cond = (const int*)d_in[1];
  float*       out  = (float*)d_out;

  // workspace: bf16 concat + two ping-pong hidden buffers (max h = 1431 -> ldh 1432)
  char* ws = (char*)d_ws;
  __bf16* concat = (__bf16*)ws;
  size_t off = (size_t)BATCH * DIN * sizeof(__bf16);          // 540672 B, 16B aligned
  const int LDH_MAX = 1432;
  __bf16* hA = (__bf16*)(ws + off);
  off += (size_t)BATCH * LDH_MAX * sizeof(__bf16);
  __bf16* hB = (__bf16*)(ws + off);

  prep_concat_kernel<<<dim3(BATCH), dim3(264), 0, stream>>>(x, cond, concat);

  int colOff = 0;
  for (int c = 0; c < 23; ++c) {
    int out_c = SIZES[c];
    int h   = isqrt_floor((long)out_c * 256);   // hidden width, matches int(sqrt)
    int ldh = (h + 7) & ~7;                     // bf16 rows stay 16B aligned

    const float* W0 = (const float*)d_in[2 + c * 6 + 0];
    const float* b0 = (const float*)d_in[2 + c * 6 + 1];
    const float* W1 = (const float*)d_in[2 + c * 6 + 2];
    const float* b1 = (const float*)d_in[2 + c * 6 + 3];
    const float* W2 = (const float*)d_in[2 + c * 6 + 4];
    const float* b2 = (const float*)d_in[2 + c * 6 + 5];

    dim3 blk(256);
    dim3 gh(BATCH / BM, (unsigned)((ldh + BN - 1) / BN));
    // layer 0: [1024 x 264] @ [264 x h] -> hA (bf16, ReLU)
    mlp_gemm_kernel<false><<<gh, blk, 0, stream>>>(concat, DIN, DIN, W0, b0,
                                                   h, ldh, hA, nullptr, 0, 0);
    // layer 1: [1024 x h] @ [h x h] -> hB (bf16, ReLU)
    mlp_gemm_kernel<false><<<gh, blk, 0, stream>>>(hA, ldh, h, W1, b1,
                                                   h, ldh, hB, nullptr, 0, 0);
    // layer 2: [1024 x h] @ [h x out_c] -> sigmoid -> d_out[:, colOff:colOff+out_c]
    dim3 go(BATCH / BM, (unsigned)((out_c + BN - 1) / BN));
    mlp_gemm_kernel<true><<<go, blk, 0, stream>>>(hB, ldh, h, W2, b2,
                                                  out_c, 0, nullptr, out,
                                                  OUT_TOTAL, colOff);
    colOff += out_c;
  }

  (void)in_sizes; (void)n_in; (void)out_size; (void)ws_size;
}
